// CodeGenAttention_14233521618985
// MI455X (gfx1250) — compile-verified, matrix operand order checked by reading, not run on hardware
//
#include <hip/hip_runtime.h>
#include <hip/hip_bf16.h>
#include <stdint.h>

typedef __bf16 bf16;
typedef __attribute__((ext_vector_type(16))) __bf16 bf16x16;
typedef __attribute__((ext_vector_type(8)))  __bf16 bf16x8;
typedef __attribute__((ext_vector_type(8)))  float  f32x8;
typedef __attribute__((ext_vector_type(4)))  int    i32x4;

#define AS1 __attribute__((address_space(1)))
#define AS3 __attribute__((address_space(3)))
// Builtin wants: (v4i AS1* global_src, v4i AS3* lds_dst, imm offset, imm cpol).
// Integer round-trips are always-valid reinterpret casts; AS3 pointers are
// 32-bit and the low 32 bits of a generic LDS address are the LDS offset.
#define TO_GA1(p)  ((AS1 i32x4*)(uintptr_t)(p))
#define TO_LDS3(p) ((AS3 i32x4*)(uint32_t)(uintptr_t)(p))

#if defined(__AMDGCN__) && \
    __has_builtin(__builtin_amdgcn_global_load_async_to_lds_b128) && \
    __has_builtin(__builtin_amdgcn_s_wait_asynccnt)
#define USE_ASYNC 1
#else
#define USE_ASYNC 0
#endif

// ---------------------------------------------------------------------------
// WMMA helper: D = A(16x32 bf16) * B(32x16 bf16) + C(16x16 f32), wave32.
// ---------------------------------------------------------------------------
__device__ __forceinline__ f32x8 wmma_bf16(bf16x16 a, bf16x16 b, f32x8 c) {
  return __builtin_amdgcn_wmma_f32_16x16x32_bf16(
      /*neg_a=*/false, a, /*neg_b=*/false, b,
      /*c_mod=*/(short)0, c, /*reuse_a=*/false, /*reuse_b=*/false);
}

// 16-bit A/B fragment from bf16 memory: two 8-elem (16B) chunks at p, p+16.
// Caller offsets p by (lane%16) rows and (lane/16)*8 K-elems, per ISA layout:
//   lanes 0-15 : K 0-7  / K 16-23 ; lanes 16-31: K 8-15 / K 24-31
__device__ __forceinline__ bf16x16 load_frag(const bf16* p) {
  bf16x8 lo = *(const bf16x8*)(p);
  bf16x8 hi = *(const bf16x8*)(p + 16);
  bf16x16 r;
#pragma unroll
  for (int i = 0; i < 8; ++i) { r[i] = lo[i]; r[i + 8] = hi[i]; }
  return r;
}

// Same fragment but from an fp32 LDS tile: convert at read (v_cvt_pk_bf16_f32).
__device__ __forceinline__ bf16x16 frag_from_f32(const float* p) {
  float4 a0 = *(const float4*)(p);
  float4 a1 = *(const float4*)(p + 4);
  float4 b0 = *(const float4*)(p + 16);
  float4 b1 = *(const float4*)(p + 20);
  bf16x16 r;
  r[0]  = (bf16)a0.x; r[1]  = (bf16)a0.y; r[2]  = (bf16)a0.z; r[3]  = (bf16)a0.w;
  r[4]  = (bf16)a1.x; r[5]  = (bf16)a1.y; r[6]  = (bf16)a1.z; r[7]  = (bf16)a1.w;
  r[8]  = (bf16)b0.x; r[9]  = (bf16)b0.y; r[10] = (bf16)b0.z; r[11] = (bf16)b0.w;
  r[12] = (bf16)b1.x; r[13] = (bf16)b1.y; r[14] = (bf16)b1.z; r[15] = (bf16)b1.w;
  return r;
}

// ---------------------------------------------------------------------------
// GEMM: C[M,N] (f32, row-major) = A[M,K] (f32, row-major) * B[N,K]^T (f32).
// Double-buffered fp32 tiles in LDS; on gfx1250 the tiles are fetched with
// GLOBAL_LOAD_ASYNC_TO_LDS_B128 (ASYNCcnt) overlapping the WMMA work on the
// other buffer; fp32->bf16 conversion happens at fragment build.
// Block: 256 threads = 8 waves (2x4), tile 128x128, K-step 32.
// ---------------------------------------------------------------------------
#define GTM 128
#define GTN 128
#define GTK 32
#define LDSKF 36  // 32 + 4 pad floats (144B row stride, 16B aligned)

__global__ __launch_bounds__(256) void gemm_bf16_wmma(
    const float* __restrict__ A, const float* __restrict__ B,
    float* __restrict__ C, int M, int N, int K) {
  __shared__ float As[2][GTM][LDSKF];
  __shared__ float Bs[2][GTN][LDSKF];

  const int tid  = threadIdx.x;
  const int wave = tid >> 5;
  const int lane = tid & 31;
  const int lh   = lane >> 4;   // lane half (K-chunk select)
  const int ll   = lane & 15;   // row/col within tile
  const int wm   = wave >> 2;   // 0..1 -> 64 rows each
  const int wn   = wave & 3;    // 0..3 -> 32 cols each
  const int blockM = blockIdx.y * GTM;
  const int blockN = blockIdx.x * GTN;

  f32x8 acc[4][2];
#pragma unroll
  for (int i = 0; i < 4; ++i)
#pragma unroll
    for (int j = 0; j < 2; ++j) acc[i][j] = (f32x8)(0.0f);

  // Stage one 128x32 fp32 tile of A and of B into LDS buffer `buf`.
  // 1024 float4 per tile / 256 threads = 4 each.
  auto stage = [&](int k0, int buf) {
#pragma unroll
    for (int it = 0; it < 4; ++it) {
      const int idx = tid + it * 256;
      const int row = idx >> 3;
      const int kc  = (idx & 7) << 2;
      const float* ga = A + (size_t)(blockM + row) * K + k0 + kc;
      const float* gb = B + (size_t)(blockN + row) * K + k0 + kc;
#if USE_ASYNC
      __builtin_amdgcn_global_load_async_to_lds_b128(
          TO_GA1(ga), TO_LDS3(&As[buf][row][kc]), 0, 0);
      __builtin_amdgcn_global_load_async_to_lds_b128(
          TO_GA1(gb), TO_LDS3(&Bs[buf][row][kc]), 0, 0);
#else
      *(float4*)&As[buf][row][kc] = *(const float4*)ga;
      *(float4*)&Bs[buf][row][kc] = *(const float4*)gb;
#endif
    }
  };

  stage(0, 0);
#if USE_ASYNC
  __builtin_amdgcn_s_wait_asynccnt(0);
#endif
  __syncthreads();

  const int ksteps = K / GTK;
  for (int ks = 0; ks < ksteps; ++ks) {
    const int cur = ks & 1;
    // Prefetch next tile into the other buffer while computing this one.
    if (ks + 1 < ksteps) stage((ks + 1) * GTK, cur ^ 1);

    bf16x16 af[4];
#pragma unroll
    for (int i = 0; i < 4; ++i)
      af[i] = frag_from_f32(&As[cur][wm * 64 + i * 16 + ll][lh * 8]);
    bf16x16 bfr[2];
#pragma unroll
    for (int j = 0; j < 2; ++j)
      bfr[j] = frag_from_f32(&Bs[cur][wn * 32 + j * 16 + ll][lh * 8]);

#pragma unroll
    for (int i = 0; i < 4; ++i)
#pragma unroll
      for (int j = 0; j < 2; ++j)
        acc[i][j] = wmma_bf16(af[i], bfr[j], acc[i][j]);

#if USE_ASYNC
    __builtin_amdgcn_s_wait_asynccnt(0);
#endif
    __syncthreads();  // includes s_wait_dscnt 0: reads done before rewrite
  }

  // C-tile layout: lane holds column ll; VGPR r holds row r + 8*lh.
#pragma unroll
  for (int i = 0; i < 4; ++i) {
#pragma unroll
    for (int j = 0; j < 2; ++j) {
      const int col = blockN + wn * 32 + j * 16 + ll;
#pragma unroll
      for (int r = 0; r < 8; ++r) {
        const int row = blockM + wm * 64 + i * 16 + lh * 8 + r;
        C[(size_t)row * N + col] = acc[i][j][r];
      }
    }
  }
}

// ---------------------------------------------------------------------------
// RoPE + scatter.  qkv[s, mp*1536 + {0:q, 512:v, 1024:k} + sub*128 + d]
// (reference slices in q, v, k order).  GPT-J rotate-every-two RoPE on d<64.
// Writes Q,K as bf16 [H][S][D] and V transposed as bf16 [H][D][S].
// ---------------------------------------------------------------------------
#define SEQ 2048
#define NH  16
#define DH  128

__global__ __launch_bounds__(128) void rope_scatter(
    const float* __restrict__ qkv, bf16* __restrict__ Qo,
    bf16* __restrict__ Ko, bf16* __restrict__ Vt) {
  const int s = blockIdx.x;
  const int h = blockIdx.y;
  const int d = threadIdx.x;
  const int mp  = h >> 2;
  const int sub = h & 3;
  const size_t base = (size_t)s * 6144 + (size_t)mp * 1536 + sub * 128;

  float q = qkv[base + d];
  float v = qkv[base + 512 + d];
  float k = qkv[base + 1024 + d];

  if (d < 64) {
    const int i = d >> 1;
    const float inv_freq = powf(10000.0f, -((float)(2 * i)) / 64.0f);
    const float ang = (float)s * inv_freq;
    float sn, cs;
    sincosf(ang, &sn, &cs);
    const float qp = qkv[base + (d ^ 1)];
    const float kp = qkv[base + 1024 + (d ^ 1)];
    if (d & 1) { q = q * cs + qp * sn; k = k * cs + kp * sn; }
    else       { q = q * cs - qp * sn; k = k * cs - kp * sn; }
  }

  const size_t qi = ((size_t)h * SEQ + s) * DH + d;
  Qo[qi] = (bf16)q;
  Ko[qi] = (bf16)k;
  Vt[((size_t)h * DH + d) * SEQ + s] = (bf16)v;
}

// ---------------------------------------------------------------------------
// Flash attention, one wave per 16 query rows, kv blocks of 32.
// Q,K: bf16 [H][S][D]; Vt: bf16 [H][D][S]; O: f32 [S][H*D].
// S = QK^T/sqrt(D) via 8 WMMA/blk; online softmax in C-layout (row stats via
// __shfl_xor within 16-lane halves); P bounced through LDS to A-fragment
// layout; O += P*V via 8 WMMA/blk.  K/V are L2-resident per head (1MB);
// global_prefetch_b8 hints pull the next kv block while this one computes.
// ---------------------------------------------------------------------------
__global__ __launch_bounds__(128) void attn_flash(
    const bf16* __restrict__ Q, const bf16* __restrict__ Kh,
    const bf16* __restrict__ Vt, float* __restrict__ O) {
  __shared__ bf16 Ps[4][16 * 40];  // per-wave P tile: 16 rows x (32+8 pad)

  const int tid  = threadIdx.x;
  const int wave = tid >> 5;
  const int lane = tid & 31;
  const int lh   = lane >> 4;
  const int ll   = lane & 15;
  const int h    = blockIdx.y;
  const int qbase = blockIdx.x * 64 + wave * 16;

  const bf16* Qp = Q  + (size_t)h * SEQ * DH;
  const bf16* Kp = Kh + (size_t)h * SEQ * DH;
  const bf16* Vp = Vt + (size_t)h * DH * SEQ;

  // Q fragments: 16 rows x 128 d, 4 K-steps of 32.
  bf16x16 qf[4];
#pragma unroll
  for (int kd = 0; kd < 4; ++kd)
    qf[kd] = load_frag(Qp + (size_t)(qbase + ll) * DH + kd * 32 + lh * 8);

  f32x8 o[8];
#pragma unroll
  for (int n = 0; n < 8; ++n) o[n] = (f32x8)(0.0f);
  float m[8], lsum[8];
#pragma unroll
  for (int r = 0; r < 8; ++r) { m[r] = -3.0e38f; lsum[r] = 0.0f; }

  const float scale = 0.08838834764831845f;  // 1/sqrt(128)

  for (int kvb = 0; kvb < qbase + 16; kvb += 32) {
    // Prefetch next kv block (K rows + V columns) into cache.
    if (kvb + 32 < qbase + 16) {
      __builtin_prefetch(Kp + (size_t)(kvb + 32 + lane) * DH, 0, 1);
      __builtin_prefetch(Vp + (size_t)(lane * 4) * SEQ + kvb + 32, 0, 1);
    }

    // ---- S = Q K^T for 16 x 32 block ----
    f32x8 st[2];
    st[0] = (f32x8)(0.0f); st[1] = (f32x8)(0.0f);
#pragma unroll
    for (int t = 0; t < 2; ++t)
#pragma unroll
      for (int kd = 0; kd < 4; ++kd) {
        bf16x16 kf = load_frag(Kp + (size_t)(kvb + t * 16 + ll) * DH + kd * 32 + lh * 8);
        st[t] = wmma_bf16(qf[kd], kf, st[t]);
      }

    // ---- scale + causal mask ----
#pragma unroll
    for (int t = 0; t < 2; ++t) {
      const int col = kvb + t * 16 + ll;
#pragma unroll
      for (int r = 0; r < 8; ++r) {
        const int row = qbase + lh * 8 + r;
        const float sv = st[t][r] * scale;
        st[t][r] = (col <= row) ? sv : -1.0e9f;
      }
    }

    // ---- row max across 32 columns (lane-halves hold distinct rows) ----
    float rn[8];
#pragma unroll
    for (int r = 0; r < 8; ++r) rn[r] = fmaxf(st[0][r], st[1][r]);
#pragma unroll
    for (int off = 1; off < 16; off <<= 1)
#pragma unroll
      for (int r = 0; r < 8; ++r)
        rn[r] = fmaxf(rn[r], __shfl_xor(rn[r], off, 16));

    float sc[8];
#pragma unroll
    for (int r = 0; r < 8; ++r) {
      const float mn = fmaxf(m[r], rn[r]);
      sc[r] = __expf(m[r] - mn);
      m[r] = mn;
      lsum[r] *= sc[r];
    }
#pragma unroll
    for (int n = 0; n < 8; ++n)
#pragma unroll
      for (int r = 0; r < 8; ++r) o[n][r] *= sc[r];

    // ---- P = exp(S - m), row-sum ----
    float ps[8];
#pragma unroll
    for (int r = 0; r < 8; ++r) {
      const float p0 = __expf(st[0][r] - m[r]);
      const float p1 = __expf(st[1][r] - m[r]);
      st[0][r] = p0; st[1][r] = p1; ps[r] = p0 + p1;
    }
#pragma unroll
    for (int off = 1; off < 16; off <<= 1)
#pragma unroll
      for (int r = 0; r < 8; ++r) ps[r] += __shfl_xor(ps[r], off, 16);
#pragma unroll
    for (int r = 0; r < 8; ++r) lsum[r] += ps[r];

    // ---- C-layout -> A-fragment via LDS bounce (same wave, dscnt-ordered) ----
    bf16* pw = &Ps[wave][0];
#pragma unroll
    for (int t = 0; t < 2; ++t)
#pragma unroll
      for (int r = 0; r < 8; ++r)
        pw[(lh * 8 + r) * 40 + t * 16 + ll] = (bf16)st[t][r];

    bf16x16 pf = load_frag(&Ps[wave][ll * 40 + lh * 8]);

    // ---- O += P * V ----
#pragma unroll
    for (int n = 0; n < 8; ++n) {
      bf16x16 vf = load_frag(Vp + (size_t)(n * 16 + ll) * SEQ + kvb + lh * 8);
      o[n] = wmma_bf16(pf, vf, o[n]);
    }
  }

  // ---- normalize and store: O[s, h*128 + d], f32 ----
  float inv[8];
#pragma unroll
  for (int r = 0; r < 8; ++r) inv[r] = 1.0f / lsum[r];
#pragma unroll
  for (int n = 0; n < 8; ++n) {
    const int col = h * DH + n * 16 + ll;
#pragma unroll
    for (int r = 0; r < 8; ++r) {
      const int row = qbase + lh * 8 + r;
      O[(size_t)row * (NH * DH) + col] = o[n][r] * inv[r];
    }
  }
}

// ---------------------------------------------------------------------------
// Host-side orchestration.
// Workspace layout (bytes):
//   [0,        50331648)  qkv f32 [2048][6144]  -> reused as attn f32 [2048][2048]
//   [50331648, 58720256)  Q  bf16 [16][2048][128]
//   [58720256, 67108864)  K  bf16 [16][2048][128]
//   [67108864, 75497472)  V^T bf16 [16][128][2048]
// ---------------------------------------------------------------------------
extern "C" void kernel_launch(void* const* d_in, const int* in_sizes, int n_in,
                              void* d_out, int out_size, void* d_ws, size_t ws_size,
                              hipStream_t stream) {
  const float* hidden = (const float*)d_in[0];
  const float* w_qkv  = (const float*)d_in[1];
  const float* w_out  = (const float*)d_in[2];
  float* out = (float*)d_out;

  char* ws = (char*)d_ws;
  float* qkv  = (float*)ws;
  bf16*  Qb   = (bf16*)(ws + 50331648ull);
  bf16*  Kb   = (bf16*)(ws + 58720256ull);
  bf16*  Vt   = (bf16*)(ws + 67108864ull);
  float* attn = qkv;  // qkv fully consumed by rope_scatter before attn writes

  // 1) qkv = hidden @ w_qkv^T : M=2048, N=6144, K=2048
  gemm_bf16_wmma<<<dim3(6144 / GTN, 2048 / GTM), 256, 0, stream>>>(
      hidden, w_qkv, qkv, 2048, 6144, 2048);

  // 2) RoPE + split/scatter to bf16 head-major Q, K, V^T
  rope_scatter<<<dim3(SEQ, NH), DH, 0, stream>>>(qkv, Qb, Kb, Vt);

  // 3) causal flash attention -> attn f32 [2048][2048]
  attn_flash<<<dim3(SEQ / 64, NH), 128, 0, stream>>>(Qb, Kb, Vt, attn);

  // 4) out = attn @ w_out^T : M=N=K=2048
  gemm_bf16_wmma<<<dim3(2048 / GTN, 2048 / GTM), 256, 0, stream>>>(
      attn, w_out, out, 2048, 2048, 2048);
}
